// UpConvolution_49374944035302
// MI455X (gfx1250) — matrix operand order repaired and checked by reading, compiled, not run on hardware
//
#include <hip/hip_runtime.h>
#include <hip/hip_bf16.h>

typedef __attribute__((ext_vector_type(16))) _Float16 v16h;
typedef __attribute__((ext_vector_type(8)))  _Float16 v8h;
typedef __attribute__((ext_vector_type(8)))  float    v8f;

#define B_    16
#define CIN   64
#define COUT  128
#define H_    64
#define W_    64
#define HW    (H_*W_)        // 4096
#define KK    9
#define KTOT  (CIN*KK)       // 576
#define HO    129
#define WO    129
#define NSTAT (B_*HO*WO)     // 266256
#define NBLK_UP (B_*33)      // 528 k_up blocks

// ---------------- workspace layout (bytes) ----------------
constexpr size_t OFF_OFF    = 0;                         // offsets f32: 16*18*4096*4 = 4718592
constexpr size_t WDEF16_OFF = 4718592;                   // w_def f16 [128][576]      = 147456
constexpr size_t BWS_OFF    = WDEF16_OFF + 147456;       // w_up  f16 [9][128][128]   = 294912
constexpr size_t FEATS_OFF  = BWS_OFF + 294912;          // feats f16 NHWC            = 16777216
constexpr size_t SUMS_OFF   = FEATS_OFF + 16777216;      // sum[128], sumsq[128]      = 1024
constexpr size_t Y_OFF      = SUMS_OFF + 1024;           // y f32 NHWC                = 136323072
constexpr size_t PART_OFF   = Y_OFF + 136323072;         // partials [2][528][128] f32 = 540672

// ---------------- gfx1250 async global->LDS helpers ----------------
#if defined(__has_builtin)
#if __has_builtin(__builtin_amdgcn_global_load_async_to_lds_b128)
#define HAVE_ASYNC_LDS 1
#endif
#endif

typedef int v4i_ __attribute__((vector_size(16)));
typedef __attribute__((address_space(1))) v4i_* as1_v4i_p;
typedef __attribute__((address_space(3))) v4i_* as3_v4i_p;

__device__ __forceinline__ void async_copy16(const void* g, void* l) {
#ifdef HAVE_ASYNC_LDS
  __builtin_amdgcn_global_load_async_to_lds_b128(
      (as1_v4i_p)(unsigned long long)g,
      (as3_v4i_p)(unsigned)(unsigned long long)l,
      0, 0);
#else
  *(v8h*)l = *(const v8h*)g;
#endif
}

__device__ __forceinline__ void async_fence() {
#ifdef HAVE_ASYNC_LDS
#if __has_builtin(__builtin_amdgcn_s_wait_asynccnt)
  __builtin_amdgcn_s_wait_asynccnt(0);
#else
  asm volatile("s_wait_asynccnt 0" ::: "memory");
#endif
#endif
}

// ---------------- kernel 0: weight repack + stat zero ----------------
__global__ __launch_bounds__(256) void k_prep(const float* __restrict__ wdef,
                                              const float* __restrict__ wup,
                                              _Float16* __restrict__ wdef16,
                                              _Float16* __restrict__ bws,
                                              float* __restrict__ sums) {
  int i = blockIdx.x * 256 + threadIdx.x;
  if (i < COUT * KTOT) wdef16[i] = (_Float16)wdef[i];   // [co][ci*9+kk], natural layout
  if (i < 9 * COUT * COUT) {
    int ci = i & 127;
    int co = (i >> 7) & 127;
    int t  = i >> 14;                                   // tap = ky*3+kx
    bws[i] = (_Float16)wup[(size_t)(ci * COUT + co) * 9 + t];
  }
  if (i < 2 * COUT) sums[i] = 0.f;
}

// ---------------- kernel 1: 3x3 offset conv (f32) ----------------
__global__ __launch_bounds__(256) void k_off(const float* __restrict__ x,
                                             const float* __restrict__ woff,
                                             const float* __restrict__ boff,
                                             float* __restrict__ off) {
  __shared__ float wl[18 * KTOT];                       // 41.5 KB
  for (int i = threadIdx.x; i < 18 * KTOT; i += 256) wl[i] = woff[i];
  __syncthreads();
  int pix = blockIdx.x * 256 + threadIdx.x;             // 65536 pixels total
  int b = pix >> 12, rem = pix & 4095;
  int y = rem >> 6, xx = rem & 63;
  float acc[18];
#pragma unroll
  for (int ch = 0; ch < 18; ++ch) acc[ch] = boff[ch];
  const float* xb = x + (size_t)b * CIN * HW;
  for (int ci = 0; ci < CIN; ++ci) {
    const float* xc = xb + (size_t)ci * HW;
    float v[9];
#pragma unroll
    for (int t = 0; t < 9; ++t) {
      int yy = y + t / 3 - 1, xx2 = xx + t % 3 - 1;
      v[t] = (yy >= 0 && yy < H_ && xx2 >= 0 && xx2 < W_) ? xc[yy * W_ + xx2] : 0.f;
    }
#pragma unroll
    for (int ch = 0; ch < 18; ++ch) {
      const float* wp = &wl[(ch * CIN + ci) * 9];
      float s = 0.f;
#pragma unroll
      for (int t = 0; t < 9; ++t) s += v[t] * wp[t];
      acc[ch] += s;
    }
  }
#pragma unroll
  for (int ch = 0; ch < 18; ++ch)
    off[((size_t)b * 18 + ch) * HW + rem] = acc[ch];
}

// ---------------- kernel 2: deformable conv via WMMA ----------------
// block = 256 threads (8 waves) handles 32 pixels x 128 Cout
__global__ __launch_bounds__(256) void k_def(const float* __restrict__ x,
                                             const float* __restrict__ off,
                                             const _Float16* __restrict__ wdef16,
                                             const float* __restrict__ bdef,
                                             _Float16* __restrict__ feats) {
  __shared__ _Float16 smem[32 * KTOT];                  // 36 KB sampled tile
  int pix0 = blockIdx.x * 32;
  int b = pix0 >> 12;
  const float* xb = x + (size_t)b * CIN * HW;

  // ---- phase 1: bilinear sampling into LDS: smem[pl][ci*9+kk] ----
  for (int pair = threadIdx.x; pair < 32 * KK; pair += 256) {
    int kk = pair >> 5, pl = pair & 31;
    int rem = (pix0 + pl) & 4095;
    int y = rem >> 6, xx = rem & 63;
    float dy = off[((size_t)b * 18 + 2 * kk) * HW + rem];
    float dx = off[((size_t)b * 18 + 2 * kk + 1) * HW + rem];
    float py = (float)(y - 1 + kk / 3) + dy;
    float px = (float)(xx - 1 + kk % 3) + dx;
    float y0f = floorf(py), x0f = floorf(px);
    float ly = py - y0f, lx = px - x0f;
    int y0 = (int)y0f, x0 = (int)x0f;
    int y1 = y0 + 1, x1 = x0 + 1;
    float w00 = (1.f - ly) * (1.f - lx), w01 = (1.f - ly) * lx;
    float w10 = ly * (1.f - lx), w11 = ly * lx;
    bool vy0 = (y0 >= 0) & (y0 < H_), vy1 = (y1 >= 0) & (y1 < H_);
    bool vx0 = (x0 >= 0) & (x0 < W_), vx1 = (x1 >= 0) & (x1 < W_);
    w00 *= (float)(vy0 && vx0); w01 *= (float)(vy0 && vx1);
    w10 *= (float)(vy1 && vx0); w11 *= (float)(vy1 && vx1);
    int cy0 = min(max(y0, 0), H_ - 1), cy1 = min(max(y1, 0), H_ - 1);
    int cx0 = min(max(x0, 0), W_ - 1), cx1 = min(max(x1, 0), W_ - 1);
    int i00 = cy0 * W_ + cx0, i01 = cy0 * W_ + cx1;
    int i10 = cy1 * W_ + cx0, i11 = cy1 * W_ + cx1;
    _Float16* sp = &smem[pl * KTOT + kk];
    const float* xc = xb;
    for (int c = 0; c < CIN; ++c) {
      float v = w00 * xc[i00] + w01 * xc[i01] + w10 * xc[i10] + w11 * xc[i11];
      sp[c * KK] = (_Float16)v;
      xc += HW;
    }
  }
  __syncthreads();

  // ---- phase 2: GEMM [16 pix x 576] x [576 x 32 cout] per wave ----
  int wv = threadIdx.x >> 5, lane = threadIdx.x & 31;
  int ln = lane & 15, hi = lane >> 4;
  int psub = wv & 1;
  int nt0 = (wv >> 1) * 2;
  v8f acc0 = {}; v8f acc1 = {};
  const _Float16* sa = &smem[(psub * 16 + ln) * KTOT];
  const _Float16* bp0 = wdef16 + (size_t)(nt0 * 16 + ln) * KTOT;
  const _Float16* bp1 = bp0 + (size_t)16 * KTOT;
  for (int kc = 0; kc < 18; ++kc) {
    int g = kc * 32 + (hi ? 16 : 0);
    v8h alo = *(const v8h*)(sa + g);
    v8h ahi = *(const v8h*)(sa + g + 8);
    v16h a;
#pragma unroll
    for (int i = 0; i < 8; ++i) { a[i] = alo[i]; a[8 + i] = ahi[i]; }
    int kb = kc * 32 + (hi ? 16 : 0);
    v16h bb0 = *(const v16h*)(bp0 + kb);
    v16h bb1 = *(const v16h*)(bp1 + kb);
    acc0 = __builtin_amdgcn_wmma_f32_16x16x32_f16(false, a, false, bb0, (short)0, acc0, false, false);
    acc1 = __builtin_amdgcn_wmma_f32_16x16x32_f16(false, a, false, bb1, (short)0, acc1, false, false);
  }
  float bu0 = bdef[nt0 * 16 + ln];
  float bu1 = bdef[(nt0 + 1) * 16 + ln];
#pragma unroll
  for (int r = 0; r < 8; ++r) {
    int pl = psub * 16 + r + (hi ? 8 : 0);
    size_t base = (size_t)(pix0 + pl) * COUT;
    feats[base + nt0 * 16 + ln] = (_Float16)(acc0[r] + bu0);
    feats[base + (nt0 + 1) * 16 + ln] = (_Float16)(acc1[r] + bu1);
  }
}

// ---------------- kernel 3: stride-2 transpose conv via WMMA + BN partials ----------------
// grid = 16*33 blocks; wave = one cout-tile; each wave: 4 output rows x 9 ox-tiles.
// Input rows needed by the whole block: iy in {2g-1, 2g, 2g+1} -> staged in LDS once
// (async global->LDS path), with a zeroed LDS pad for parity/range-invalid A rows so
// all A-fragment ds_loads are unconditional.
__global__ __launch_bounds__(256) void k_up(const _Float16* __restrict__ feats,
                                            const _Float16* __restrict__ bws,
                                            const float* __restrict__ bup,
                                            float* __restrict__ y,
                                            float* __restrict__ part) {
  __shared__ _Float16 frow[3 * 64 * COUT + 128];        // 3 staged rows + 256B zero pad = 49408 B
  int blk = blockIdx.x;
  int b = blk / 33, oyg = blk % 33;
  int iy0 = 2 * oyg - 1;                                // slot s -> iy0 + s

  // zero pad block (128 halves = 64 floats)
  if (threadIdx.x < 64) ((float*)(frow + 3 * 64 * COUT))[threadIdx.x] = 0.f;

  // stage up to 3 feats rows (16 KB each) into LDS
  const _Float16* fbase = feats + (size_t)b * HW * COUT;
  for (int c = threadIdx.x; c < 3 * 1024; c += 256) {   // 1024 x 16B chunks per row
    int s = c >> 10, off16 = c & 1023;
    int iy = iy0 + s;
    if (iy >= 0 && iy < H_) {
      const _Float16* g = fbase + ((size_t)iy * W_ * COUT) + (size_t)off16 * 8;
      _Float16* l = frow + s * (W_ * COUT) + off16 * 8;
      async_copy16(g, l);
    }
  }
  async_fence();
  __syncthreads();

  int wv = threadIdx.x >> 5, lane = threadIdx.x & 31;
  int ln = lane & 15, hi = lane >> 4;
  int co = wv * 16 + ln;
  float bupc = bup[co];
  float ssum = 0.f, ssq = 0.f;
  const _Float16* zpad = frow + 3 * 64 * COUT;

  for (int r4 = 0; r4 < 4; ++r4) {
    int oy = oyg * 4 + r4;
    if (oy >= HO) break;
    v8f acc[9] = {};
    for (int ky = 0; ky < 3; ++ky) {
      int t2 = oy - ky;
      if (t2 < 0 || (t2 & 1)) continue;
      int iy = t2 >> 1;
      if (iy >= H_) continue;
      const _Float16* lrow = frow + (iy - iy0) * (W_ * COUT);
      for (int kx = 0; kx < 3; ++kx) {
        const _Float16* brow = bws + ((size_t)(ky * 3 + kx) * COUT + co) * COUT;
        v16h bf[4];
#pragma unroll
        for (int kc = 0; kc < 4; ++kc)
          bf[kc] = *(const v16h*)(brow + kc * 32 + (hi ? 16 : 0));
#pragma unroll
        for (int oxt = 0; oxt < 9; ++oxt) {
          int ox = oxt * 16 + ln;                       // A-row M = lane%16
          int t3 = ox - kx;
          bool vld = (ox < WO) && (t3 >= 0) && !(t3 & 1) && ((t3 >> 1) < W_);
          const _Float16* ap = vld ? (lrow + (size_t)(t3 >> 1) * COUT) : zpad;
#pragma unroll
          for (int kc = 0; kc < 4; ++kc) {
            int g = kc * 32 + (hi ? 16 : 0);
            v8h alo = *(const v8h*)(ap + g);            // unconditional ds_load_b128
            v8h ahi = *(const v8h*)(ap + g + 8);
            v16h a;
#pragma unroll
            for (int i = 0; i < 8; ++i) { a[i] = alo[i]; a[8 + i] = ahi[i]; }
            acc[oxt] = __builtin_amdgcn_wmma_f32_16x16x32_f16(false, a, false, bf[kc], (short)0, acc[oxt], false, false);
          }
        }
      }
    }
    // store y (NHWC) + accumulate per-channel stats
    float* yrow = y + ((size_t)(b * HO + oy) * WO) * COUT;
#pragma unroll
    for (int oxt = 0; oxt < 9; ++oxt) {
#pragma unroll
      for (int r = 0; r < 8; ++r) {
        int ox = oxt * 16 + r + (hi ? 8 : 0);           // D-row M = r + 8*hi
        if (ox < WO) {
          float v = acc[oxt][r] + bupc;
          ssum += v; ssq += v * v;
          yrow[(size_t)ox * COUT + co] = v;
        }
      }
    }
  }
  // deterministic partial reduction: combine lane pairs (same co), write per-block slot
  ssum += __shfl_down(ssum, 16);
  ssq  += __shfl_down(ssq, 16);
  if (hi == 0) {
    part[(size_t)blk * COUT + co] = ssum;
    part[(size_t)NBLK_UP * COUT + (size_t)blk * COUT + co] = ssq;
  }
}

// ---------------- kernel 3.5: fixed-order stat reduction ----------------
__global__ __launch_bounds__(256) void k_red(const float* __restrict__ part,
                                             float* __restrict__ sums) {
  int t = threadIdx.x;                                  // 256 = {sum,sq} x 128 channels
  int co = t & 127, which = t >> 7;
  const float* p = part + (size_t)which * NBLK_UP * COUT + co;
  float acc = 0.f;
  for (int blk = 0; blk < NBLK_UP; ++blk) acc += p[(size_t)blk * COUT];
  sums[which * COUT + co] = acc;
}

// ---------------- kernel 4: BN finalize + ReLU + NHWC->NCHW ----------------
__global__ __launch_bounds__(256) void k_bn(const float* __restrict__ y,
                                            const float* __restrict__ sums,
                                            const float* __restrict__ gamma,
                                            const float* __restrict__ beta,
                                            float* __restrict__ out) {
  __shared__ float tile[129 * 129];                     // [ox][co], pitch 129 to dodge bank conflicts
  __shared__ float sa[COUT], sc[COUT];
  int blk = blockIdx.x;
  int b = blk / HO, oy = blk % HO;
  const float* src = y + ((size_t)(b * HO + oy) * WO) * COUT;
  for (int i = threadIdx.x; i < WO * COUT; i += 256) {
    int ox = i >> 7, co = i & 127;
    tile[ox * 129 + co] = src[i];
  }
  if (threadIdx.x < COUT) {
    int co = threadIdx.x;
    float n = (float)NSTAT;
    float mu = sums[co] / n;
    float var = sums[COUT + co] / n - mu * mu;
    float inv = rsqrtf(var + 1e-5f);
    float a = gamma[co] * inv;
    sa[co] = a;
    sc[co] = beta[co] - mu * a;
  }
  __syncthreads();
  for (int i = threadIdx.x; i < WO * COUT; i += 256) {
    int co = i / WO, ox = i % WO;
    float v = sa[co] * tile[ox * 129 + co] + sc[co];
    out[((size_t)(b * COUT + co) * HO + oy) * WO + ox] = fmaxf(v, 0.f);
  }
}

// ---------------- launcher ----------------
extern "C" void kernel_launch(void* const* d_in, const int* in_sizes, int n_in,
                              void* d_out, int out_size, void* d_ws, size_t ws_size,
                              hipStream_t stream) {
  (void)in_sizes; (void)n_in; (void)out_size; (void)ws_size;
  const float* x     = (const float*)d_in[0];
  const float* woff  = (const float*)d_in[1];
  const float* boff  = (const float*)d_in[2];
  const float* wdef  = (const float*)d_in[3];
  const float* bdef  = (const float*)d_in[4];
  const float* wup   = (const float*)d_in[5];
  const float* bup   = (const float*)d_in[6];
  const float* gamma = (const float*)d_in[7];
  const float* beta  = (const float*)d_in[8];

  char* ws = (char*)d_ws;
  float*    off    = (float*)(ws + OFF_OFF);
  _Float16* wdef16 = (_Float16*)(ws + WDEF16_OFF);
  _Float16* bws    = (_Float16*)(ws + BWS_OFF);
  _Float16* feats  = (_Float16*)(ws + FEATS_OFF);
  float*    sums   = (float*)(ws + SUMS_OFF);
  float*    ynhwc  = (float*)(ws + Y_OFF);
  float*    part   = (float*)(ws + PART_OFF);
  float*    out    = (float*)d_out;

  k_prep<<<(9 * COUT * COUT + 255) / 256, 256, 0, stream>>>(wdef, wup, wdef16, bws, sums);
  k_off<<<(B_ * HW) / 256, 256, 0, stream>>>(x, woff, boff, off);
  k_def<<<(B_ * HW) / 32, 256, 0, stream>>>(x, off, wdef16, bdef, feats);
  k_up<<<NBLK_UP, 256, 0, stream>>>(feats, bws, bup, ynhwc, part);
  k_red<<<1, 256, 0, stream>>>(part, sums);
  k_bn<<<B_ * HO, 256, 0, stream>>>(ynhwc, sums, gamma, beta, out);
}
